// CIGHCL_Heterarchical_60687887892780
// MI455X (gfx1250) — compile-verified, matrix operand order checked by reading, not run on hardware
//
#include <hip/hip_runtime.h>

#define HH 128

typedef __attribute__((ext_vector_type(16))) __bf16 v16bf;
typedef __attribute__((ext_vector_type(8)))  float  v8f;

__device__ __forceinline__ unsigned short f2bf(float f) {
  unsigned int u = __float_as_uint(f);
  u += 0x7FFFu + ((u >> 16) & 1u);
  return (unsigned short)(u >> 16);
}

union FragU {
  v16bf v;
  uint4 q[2];
};

// ---------------- degree counting ----------------
__global__ void count_edges_kernel(const int* __restrict__ dst, int E, float* __restrict__ cnt) {
  int i = blockIdx.x * blockDim.x + threadIdx.x;
  if (i < E) atomicAdd(&cnt[dst[i]], 1.0f);
}

__global__ void invert_kernel(float* __restrict__ c, int N) {
  int i = blockIdx.x * blockDim.x + threadIdx.x;
  if (i < N) c[i] = 1.0f / fmaxf(c[i], 1.0f);
}

// ---------------- weight preparation -------------
// WtA = 0.5*We^T (bf16), WtB = 0.5*(W1+W2+W3)^T (bf16), bias = 0.5*(be+bs)
__global__ void prep_pair_kernel(const float* __restrict__ We, const float* __restrict__ W1,
                                 const float* __restrict__ W2, const float* __restrict__ W3,
                                 const float* __restrict__ be, const float* __restrict__ bs,
                                 unsigned short* __restrict__ WtA, unsigned short* __restrict__ WtB,
                                 float* __restrict__ bias) {
  int i = blockIdx.x * blockDim.x + threadIdx.x;
  if (i < HH * HH) {
    int k = i >> 7, n = i & 127;
    WtA[n * HH + k] = f2bf(0.5f * We[i]);
    WtB[n * HH + k] = f2bf(0.5f * (W1[i] + W2[i] + W3[i]));
  }
  if (i < HH) bias[i] = 0.5f * (be[i] + bs[i]);
}

__global__ void prep_lin_kernel(const float* __restrict__ W, unsigned short* __restrict__ Wt) {
  int i = blockIdx.x * blockDim.x + threadIdx.x;  // W row-major [k=256][n=128]
  if (i < 256 * HH) {
    int k = i >> 7, n = i & 127;
    Wt[n * 256 + k] = f2bf(W[i]);
  }
}

// ---------------- edge scatter-add ---------------
// one wave per edge; lane covers 4 consecutive floats of the 128-dim vector
__global__ void scatter_add_kernel(const float* __restrict__ xsrc, const int* __restrict__ src,
                                   const int* __restrict__ dst, int E, float* __restrict__ agg) {
  int wid  = (blockIdx.x * blockDim.x + threadIdx.x) >> 5;
  int lane = threadIdx.x & 31;
  if (wid >= E) return;
  int s = src[wid];
  int d = dst[wid];
  float4 v = ((const float4*)(xsrc + (size_t)s * HH))[lane];
  float* op = agg + (size_t)d * HH + lane * 4;
  atomicAdd(op + 0, v.x);
  atomicAdd(op + 1, v.y);
  atomicAdd(op + 2, v.z);
  atomicAdd(op + 3, v.w);
}

// ---------------- fused SAGE GEMM ----------------
// out[M x 128] = act( (A .* invA[row]) @ WtA^T + X @ WtB^T + bias )
// WtA/WtB are pre-transposed bf16 [n][k]; A/X f32 [row][k].
__global__ __launch_bounds__(256)
void sage_gemm_kernel(const float* __restrict__ A, const float* __restrict__ invA,
                      const float* __restrict__ X,
                      const unsigned short* __restrict__ WtA,
                      const unsigned short* __restrict__ WtB,
                      const float* __restrict__ bias,
                      float* __restrict__ out,
                      int M, int K, int kshift, int relu) {
  extern __shared__ __attribute__((aligned(16))) unsigned short lds[];
  const int strideK = K + 8;                  // ushort stride, 16B aligned, bank-skewed
  unsigned short* lA  = lds;
  unsigned short* lWa = lds + 128 * strideK;
  unsigned short* lX  = lds + 2 * 128 * strideK;
  unsigned short* lWb = lds + 3 * 128 * strideK;

  const int tid = threadIdx.x;
  const int rowBase = blockIdx.x * 128;
  const int kmask = K - 1;
  const int total = 128 * K;

  // stage A (row-scaled by invA) and WtA into LDS as bf16
  for (int i = tid; i < total; i += 256) {
    int r = i >> kshift, k = i & kmask;
    int gr = rowBase + r;
    float v = 0.0f;
    if (gr < M) {
      v = A[(size_t)gr * K + k];
      if (invA) v *= invA[gr];
    }
    lA[r * strideK + k]  = f2bf(v);
    lWa[r * strideK + k] = WtA[i];
  }
  if (X) {
    for (int i = tid; i < total; i += 256) {
      int r = i >> kshift, k = i & kmask;
      int gr = rowBase + r;
      float v = (gr < M) ? X[(size_t)gr * K + k] : 0.0f;
      lX[r * strideK + k]  = f2bf(v);
      lWb[r * strideK + k] = WtB[i];
    }
  }
  __syncthreads();

  const int w    = tid >> 5;      // wave id: owns 16-row strip
  const int lane = tid & 31;
  const int lr   = lane & 15;
  const int hf   = lane >> 4;

  v8f acc[8] = {};

  const int mrow = (w * 16 + lr) * strideK;
  for (int s = 0; s < K; s += 32) {
    FragU a;
    a.q[0] = *(const uint4*)(lA + mrow + s + hf * 8);
    a.q[1] = *(const uint4*)(lA + mrow + s + 16 + hf * 8);
#pragma unroll
    for (int nt = 0; nt < 8; ++nt) {
      FragU b;
      const int nrow = (nt * 16 + lr) * strideK;
      b.q[0] = *(const uint4*)(lWa + nrow + s + hf * 8);
      b.q[1] = *(const uint4*)(lWa + nrow + s + 16 + hf * 8);
      acc[nt] = __builtin_amdgcn_wmma_f32_16x16x32_bf16(
          false, a.v, false, b.v, (short)0, acc[nt], false, false);
    }
  }
  if (X) {
    for (int s = 0; s < K; s += 32) {
      FragU a;
      a.q[0] = *(const uint4*)(lX + mrow + s + hf * 8);
      a.q[1] = *(const uint4*)(lX + mrow + s + 16 + hf * 8);
#pragma unroll
      for (int nt = 0; nt < 8; ++nt) {
        FragU b;
        const int nrow = (nt * 16 + lr) * strideK;
        b.q[0] = *(const uint4*)(lWb + nrow + s + hf * 8);
        b.q[1] = *(const uint4*)(lWb + nrow + s + 16 + hf * 8);
        acc[nt] = __builtin_amdgcn_wmma_f32_16x16x32_bf16(
            false, a.v, false, b.v, (short)0, acc[nt], false, false);
      }
    }
  }

  // D layout: VGPR r -> row (r + 8*hf) of tile, lane lr -> column within n-tile
#pragma unroll
  for (int nt = 0; nt < 8; ++nt) {
    int col = nt * 16 + lr;
    float bb = bias ? bias[col] : 0.0f;
#pragma unroll
    for (int r = 0; r < 8; ++r) {
      int gr = rowBase + w * 16 + hf * 8 + r;
      if (gr < M) {
        float v = acc[nt][r] + bb;
        if (relu) v = fmaxf(v, 0.0f);
        out[(size_t)gr * HH + col] = v;
      }
    }
  }
}

// ---------------- final fusion -------------------
__global__ void fuse_kernel(const float* __restrict__ alpha,
                            const float* __restrict__ hui, const float* __restrict__ hii,
                            const float* __restrict__ hia,
                            float* __restrict__ fused, float* __restrict__ wout,
                            long long n) {
  float a0 = alpha[0], a1 = alpha[1], a2 = alpha[2];
  float m  = fmaxf(a0, fmaxf(a1, a2));
  float e0 = __expf(a0 - m), e1 = __expf(a1 - m), e2 = __expf(a2 - m);
  float inv = 1.0f / (e0 + e1 + e2);
  float w0 = e0 * inv, w1 = e1 * inv, w2 = e2 * inv;
  long long i = (long long)blockIdx.x * blockDim.x + threadIdx.x;
  if (i < n) fused[i] = w0 * hui[i] + w1 * hii[i] + w2 * hia[i];
  if (i == 0) { wout[0] = w0; wout[1] = w1; wout[2] = w2; }
}

// =================================================
static inline char* carve(char*& p, size_t bytes) {
  char* r = p;
  p += (bytes + 255) & ~(size_t)255;
  return r;
}

extern "C" void kernel_launch(void* const* d_in, const int* in_sizes, int n_in,
                              void* d_out, int out_size, void* d_ws, size_t ws_size,
                              hipStream_t stream) {
  (void)in_sizes; (void)n_in; (void)out_size; (void)ws_size;

  const int NU = 100000, NI = 50000, NA = 5000;
  const int ER = 1000000, ES = 500000, EH = 250000;
  const size_t NIH = (size_t)NI * HH, NUH = (size_t)NU * HH, NAH = (size_t)NA * HH;

  const float* x_feat   = (const float*)d_in[0];
  const float* user_emb = (const float*)d_in[1];
  const float* attr_emb = (const float*)d_in[2];
  const float* lin_W    = (const float*)d_in[3];
  const float* lin_b    = (const float*)d_in[4];
  const float* ui_Wl    = (const float*)d_in[5];
  const float* ui_bl    = (const float*)d_in[6];
  const float* ui_Wr    = (const float*)d_in[7];
  const float* ii_Wl    = (const float*)d_in[8];
  const float* ii_bl    = (const float*)d_in[9];
  const float* ii_Wr    = (const float*)d_in[10];
  const float* ia_Wl    = (const float*)d_in[11];
  const float* ia_bl    = (const float*)d_in[12];
  const float* ia_Wr    = (const float*)d_in[13];
  const float* alpha    = (const float*)d_in[14];
  const int*   e_rates  = (const int*)d_in[15];
  const int*   e_sim    = (const int*)d_in[16];
  const int*   e_has    = (const int*)d_in[17];

  const int* rates_src = e_rates;         // user ids
  const int* rates_dst = e_rates + ER;    // item ids
  const int* sim_src   = e_sim;
  const int* sim_dst   = e_sim + ES;
  const int* has_src   = e_has;           // item ids
  const int* has_dst   = e_has + EH;      // attr ids

  // ---- workspace carve
  char* p = (char*)d_ws;
  float* x_item = (float*)carve(p, NIH * 4);
  float* hu2    = (float*)carve(p, NUH * 4);
  float* hi2    = (float*)carve(p, NIH * 4);
  float* scr    = (float*)carve(p, NUH * 4);  // agg_u; later hc = scr, hv = scr+NIH
  float* agg_i  = (float*)carve(p, NIH * 4);
  float* ha2    = (float*)carve(p, NAH * 4);
  float* agg_a  = (float*)carve(p, NAH * 4);
  float* inv_ri = (float*)carve(p, (size_t)NI * 4);
  float* inv_ru = (float*)carve(p, (size_t)NU * 4);
  float* inv_s  = (float*)carve(p, (size_t)NI * 4);
  float* inv_ha = (float*)carve(p, (size_t)NA * 4);
  float* inv_hi = (float*)carve(p, (size_t)NI * 4);
  unsigned short* wbuf = (unsigned short*)carve(p, (size_t)(8 * 2 * HH * HH + 256 * HH) * 2);
  float* bbuf = (float*)carve(p, (size_t)8 * HH * 4);
  unsigned short* WtL = wbuf + (size_t)8 * 2 * HH * HH;

  float* out  = (float*)d_out;
  float* hui  = out + NIH;
  float* hii  = out + 2 * NIH;
  float* hia  = out + 3 * NIH;
  float* wout = out + 4 * NIH;

  hipFuncSetAttribute((const void*)sage_gemm_kernel,
                      hipFuncAttributeMaxDynamicSharedMemorySize, 160 * 1024);

  // ---- combined-weight prep (folds 0.5 scale + self-loop weight sum)
  auto W4 = [](const float* b, int l, int r) { return b + (size_t)(l * 4 + r) * HH * HH; };
  auto B4 = [](const float* b, int l, int r) { return b + (size_t)(l * 4 + r) * HH; };
  auto W2 = [](const float* b, int l, int r) { return b + (size_t)(l * 2 + r) * HH * HH; };
  auto B2 = [](const float* b, int l, int r) { return b + (size_t)(l * 2 + r) * HH; };

  struct OpCfg { const float *We, *W1, *W2, *W3, *be, *bs; };
  OpCfg ops[8] = {
    { W4(ui_Wl,0,0), W4(ui_Wr,0,0), W4(ui_Wl,0,3), W4(ui_Wr,0,3), B4(ui_bl,0,0), B4(ui_bl,0,3) }, // ui L0 item
    { W4(ui_Wl,0,1), W4(ui_Wr,0,1), W4(ui_Wl,0,2), W4(ui_Wr,0,2), B4(ui_bl,0,1), B4(ui_bl,0,2) }, // ui L0 user
    { W4(ui_Wl,1,0), W4(ui_Wr,1,0), W4(ui_Wl,1,3), W4(ui_Wr,1,3), B4(ui_bl,1,0), B4(ui_bl,1,3) }, // ui L1 item
    { W2(ii_Wl,0,0), W2(ii_Wr,0,0), W2(ii_Wl,0,1), W2(ii_Wr,0,1), B2(ii_bl,0,0), B2(ii_bl,0,1) }, // ii L0
    { W2(ii_Wl,1,0), W2(ii_Wr,1,0), W2(ii_Wl,1,1), W2(ii_Wr,1,1), B2(ii_bl,1,0), B2(ii_bl,1,1) }, // ii L1
    { W4(ia_Wl,0,0), W4(ia_Wr,0,0), W4(ia_Wl,0,3), W4(ia_Wr,0,3), B4(ia_bl,0,0), B4(ia_bl,0,3) }, // ia L0 attr
    { W4(ia_Wl,0,1), W4(ia_Wr,0,1), W4(ia_Wl,0,2), W4(ia_Wr,0,2), B4(ia_bl,0,1), B4(ia_bl,0,2) }, // ia L0 item
    { W4(ia_Wl,1,1), W4(ia_Wr,1,1), W4(ia_Wl,1,2), W4(ia_Wr,1,2), B4(ia_bl,1,1), B4(ia_bl,1,2) }, // ia L1 item
  };
  for (int o = 0; o < 8; ++o) {
    prep_pair_kernel<<<64, 256, 0, stream>>>(
        ops[o].We, ops[o].W1, ops[o].W2, ops[o].W3, ops[o].be, ops[o].bs,
        wbuf + (size_t)o * 2 * HH * HH, wbuf + (size_t)o * 2 * HH * HH + HH * HH,
        bbuf + (size_t)o * HH);
  }
  prep_lin_kernel<<<128, 256, 0, stream>>>(lin_W, WtL);

  auto WA = [&](int o) { return wbuf + (size_t)o * 2 * HH * HH; };
  auto WB = [&](int o) { return wbuf + (size_t)o * 2 * HH * HH + HH * HH; };

  // ---- degree counts -> reciprocal means
  auto countE = [&](const int* dst, int E, float* cnt, int N) {
    hipMemsetAsync(cnt, 0, (size_t)N * 4, stream);
    count_edges_kernel<<<(E + 255) / 256, 256, 0, stream>>>(dst, E, cnt);
    invert_kernel<<<(N + 255) / 256, 256, 0, stream>>>(cnt, N);
  };
  countE(rates_dst, ER, inv_ri, NI);
  countE(rates_src, ER, inv_ru, NU);
  countE(sim_dst,   ES, inv_s,  NI);
  countE(has_dst,   EH, inv_ha, NA);
  countE(has_src,   EH, inv_hi, NI);

  auto gemm = [&](const float* A, const float* iA, const float* X,
                  const unsigned short* Wa, const unsigned short* Wb,
                  const float* bias, float* o, int M, int K, int relu) {
    int kshift = (K == 256) ? 8 : 7;
    size_t ldsB = (size_t)(X ? 4 : 2) * 128 * (K + 8) * sizeof(unsigned short);
    sage_gemm_kernel<<<(M + 127) / 128, 256, ldsB, stream>>>(
        A, iA, X, Wa, Wb, bias, o, M, K, kshift, relu);
  };
  auto scat = [&](const float* xs, const int* si, const int* di, int E,
                  float* agg, size_t aggBytes) {
    hipMemsetAsync(agg, 0, aggBytes, stream);
    scatter_add_kernel<<<(E * 32 + 255) / 256, 256, 0, stream>>>(xs, si, di, E, agg);
  };

  // ---- x_item = x_feat @ lin_W + lin_b  (K=256, no relu)
  gemm(x_feat, nullptr, nullptr, WtL, nullptr, lin_b, x_item, NI, 256, 0);

  // ---- user-item branch
  scat(user_emb, rates_src, rates_dst, ER, agg_i, NIH * 4);          // users -> items
  scat(x_item,   rates_dst, rates_src, ER, scr,   NUH * 4);          // items -> users
  gemm(agg_i, inv_ri, x_item,   WA(0), WB(0), bbuf + 0 * HH, hi2, NI, 128, 1);
  gemm(scr,   inv_ru, user_emb, WA(1), WB(1), bbuf + 1 * HH, hu2, NU, 128, 1);
  scat(hu2, rates_src, rates_dst, ER, agg_i, NIH * 4);
  gemm(agg_i, inv_ri, hi2, WA(2), WB(2), bbuf + 2 * HH, hui, NI, 128, 1);
  // (layer-1 user update is dead in the reference -> skipped)

  // ---- item-item branch (reuse scr as hc)
  float* hc = scr;
  scat(x_item, sim_src, sim_dst, ES, agg_i, NIH * 4);
  gemm(agg_i, inv_s, x_item, WA(3), WB(3), bbuf + 3 * HH, hc, NI, 128, 1);
  scat(hc, sim_src, sim_dst, ES, agg_i, NIH * 4);
  gemm(agg_i, inv_s, hc, WA(4), WB(4), bbuf + 4 * HH, hii, NI, 128, 1);

  // ---- item-attr branch (reuse scr+NIH as hv)
  float* hv = scr + NIH;
  scat(x_item,   has_src, has_dst, EH, agg_a, NAH * 4);              // items -> attrs
  scat(attr_emb, has_dst, has_src, EH, agg_i, NIH * 4);              // attrs -> items
  gemm(agg_a, inv_ha, attr_emb, WA(5), WB(5), bbuf + 5 * HH, ha2, NA, 128, 1);
  gemm(agg_i, inv_hi, x_item,   WA(6), WB(6), bbuf + 6 * HH, hv,  NI, 128, 1);
  scat(ha2, has_dst, has_src, EH, agg_i, NIH * 4);
  gemm(agg_i, inv_hi, hv, WA(7), WB(7), bbuf + 7 * HH, hia, NI, 128, 1);
  // (layer-1 attr update is dead in the reference -> skipped)

  // ---- softmax(alpha) fusion + weight output
  fuse_kernel<<<(int)((NIH + 255) / 256), 256, 0, stream>>>(
      alpha, hui, hii, hia, out, wout, (long long)NIH);
}
